// PolaLinearAttention_27462020891316
// MI455X (gfx1250) — compile-verified, matrix operand order checked by reading, not run on hardware
//
#include <hip/hip_runtime.h>

// ---------------------------------------------------------------------------
// PolaLinearAttention fused kernel for MI455X (gfx1250, wave32, WMMA bf16)
//
//   x = LayerNorm(features)                      [48 rows per WG, H=768]
//   Q,K,V = x @ W{q,k,v}^T + b                   (bf16 WMMA, f32 accum)
//   S± = relu(±Q) relu(±K)^T / sqrt(H), softmax over T=3
//   out = (softmax(S+) - softmax(S-)) @ V
//   y = x + out @ Wo^T + bo                      (bf16 WMMA, f32 accum)
//
// Roofline: 0.93 TFLOP total vs 1.2 GB minimum HBM traffic (52 us at
// 23.3 TB/s) -> compute-bound at bf16 WMMA rates, so everything is fused:
// features read once, output written once, weights converted to bf16 once
// and streamed from L2 by every workgroup.
// ---------------------------------------------------------------------------

#define Hdim        768
#define ROWS_PER_WG 48          // 16 batches * T=3  (attention stays WG-local)
#define BATCH_PER_WG 16
#define NTHREADS    256         // 8 wave32
#define NWAVES      8
#define KSTEPS      24          // 768 / 32 (WMMA K)
#define NBLK        48          // 768 / 16 (WMMA N)
#define NUNITS      144         // 3 matrices (Q,K,V) * 48 column blocks

typedef __attribute__((ext_vector_type(16))) __bf16 v16bf;
typedef __attribute__((ext_vector_type(8)))  float  v8f;

__device__ __forceinline__ unsigned short f2bf(float f) {   // RNE f32->bf16
  unsigned u = __builtin_bit_cast(unsigned, f);
  u += 0x7FFFu + ((u >> 16) & 1u);
  return (unsigned short)(u >> 16);
}
__device__ __forceinline__ float bf2f(unsigned short h) {
  return __builtin_bit_cast(float, ((unsigned)h) << 16);
}

// Load one 16x16x32 bf16 operand fragment for this lane.
// ISA layout (A 16x32 / B 32x16, 16-bit): lanes 0-15 hold K=0..7 & 16..23,
// lanes 16-31 hold K=8..15 & 24..31 -> two 16B loads at +0 and +16 elements.
__device__ __forceinline__ v16bf load_frag(const unsigned short* p) {
  union { v16bf v; uint4 q[2]; } r;
  r.q[0] = *(const uint4*)(p);
  r.q[1] = *(const uint4*)(p + 16);
  return r.v;
}

#define BFWMMA(A, B, C) \
  __builtin_amdgcn_wmma_f32_16x16x32_bf16(false, (A), false, (B), (short)0, (C), false, false)

// ---------------------------------------------------------------------------
// Pre-pass: convert the four fp32 weight matrices to bf16 in workspace.
// Layout in ws: [Wq | Wk | Wv | Wo], each H*H, element (m, o, h) row-major.
// ---------------------------------------------------------------------------
__global__ __launch_bounds__(256)
void convert_weights_kernel(const float* __restrict__ wq, const float* __restrict__ wk,
                            const float* __restrict__ wv, const float* __restrict__ wo,
                            unsigned short* __restrict__ dst) {
  int idx = blockIdx.x * blockDim.x + threadIdx.x;
  if (idx >= 4 * Hdim * Hdim) return;
  int m = idx / (Hdim * Hdim);
  int r = idx % (Hdim * Hdim);
  const float* src = (m == 0) ? wq : (m == 1) ? wk : (m == 2) ? wv : wo;
  dst[idx] = f2bf(src[r]);
}

// ---------------------------------------------------------------------------
// Fused main kernel.
// ---------------------------------------------------------------------------
__global__ __launch_bounds__(NTHREADS, 1)
void pola_fused_kernel(const float* __restrict__ feat,
                       const unsigned short* __restrict__ wbf,
                       const float* __restrict__ bq, const float* __restrict__ bk,
                       const float* __restrict__ bv, const float* __restrict__ bo,
                       const float* __restrict__ ln_g, const float* __restrict__ ln_b,
                       float* __restrict__ out)
{
  extern __shared__ unsigned char smem[];
  unsigned short* sX = (unsigned short*)smem;             // 48*768 bf16 (LN out)
  unsigned short* sQ = sX + ROWS_PER_WG * Hdim;
  unsigned short* sK = sQ + ROWS_PER_WG * Hdim;
  unsigned short* sV = sK + ROWS_PER_WG * Hdim;           // later: attn@V in place
  float* sSP  = (float*)(sV + ROWS_PER_WG * Hdim);        // 144 pos scores
  float* sSN  = sSP + 144;                                // 144 neg scores
  float* sWgt = sSN + 144;                                // 144 combined attn weights

  const int tid  = threadIdx.x;
  const int lane = tid & 31;
  const int wave = tid >> 5;
  const int part = lane >> 4;   // which K-half this lane holds
  const int nlo  = lane & 15;   // row (A) / column (B,D) index within tile
  const size_t row0 = (size_t)blockIdx.x * ROWS_PER_WG;

  if (tid < 144) { sSP[tid] = 0.0f; sSN[tid] = 0.0f; }

  // ---- Phase 0: LayerNorm -> sX (bf16). One row per wave, 24 floats/lane.
  for (int r = wave; r < ROWS_PER_WG; r += NWAVES) {
    const float4* fr = (const float4*)(feat + (row0 + r) * Hdim);
    float4 v[6];
    float s = 0.f, ss = 0.f;
#pragma unroll
    for (int i = 0; i < 6; ++i) {
      v[i] = fr[i * 32 + lane];
      s  += v[i].x + v[i].y + v[i].z + v[i].w;
      ss += v[i].x * v[i].x + v[i].y * v[i].y + v[i].z * v[i].z + v[i].w * v[i].w;
    }
#pragma unroll
    for (int off = 16; off > 0; off >>= 1) {
      s  += __shfl_xor(s,  off, 32);
      ss += __shfl_xor(ss, off, 32);
    }
    const float inv  = 1.0f / (float)Hdim;
    const float mu   = s * inv;
    const float var  = ss * inv - mu * mu;
    const float rstd = rsqrtf(var + 1e-5f);
#pragma unroll
    for (int i = 0; i < 6; ++i) {
      const int vecidx = i * 32 + lane;
      const float4 g  = ((const float4*)ln_g)[vecidx];
      const float4 bb = ((const float4*)ln_b)[vecidx];
      const unsigned short o0 = f2bf((v[i].x - mu) * rstd * g.x + bb.x);
      const unsigned short o1 = f2bf((v[i].y - mu) * rstd * g.y + bb.y);
      const unsigned short o2 = f2bf((v[i].z - mu) * rstd * g.z + bb.z);
      const unsigned short o3 = f2bf((v[i].w - mu) * rstd * g.w + bb.w);
      uint2 pk;
      pk.x = (unsigned)o0 | ((unsigned)o1 << 16);
      pk.y = (unsigned)o2 | ((unsigned)o3 << 16);
      *(uint2*)&sX[(size_t)r * Hdim + (size_t)vecidx * 4] = pk;
    }
  }
  __syncthreads();

  // ---- Phase 1: Q,K,V = x @ W^T + b via WMMA bf16 ------------------------
  // Each wave owns (matrix, 16-column block) units; 3 M-tile accumulator
  // chains per unit hide WMMA latency.
  const unsigned short* sA0 = sX + (size_t)(nlo +  0) * Hdim;
  const unsigned short* sA1 = sX + (size_t)(nlo + 16) * Hdim;
  const unsigned short* sA2 = sX + (size_t)(nlo + 32) * Hdim;

  for (int u = wave; u < NUNITS; u += NWAVES) {
    const int m   = u / NBLK;                 // 0=Q 1=K 2=V
    const int nb  = u % NBLK;
    const int col = nb * 16 + nlo;
    const unsigned short* wcol = wbf + ((size_t)m * Hdim + col) * Hdim;
    const float* bias   = (m == 0) ? bq : (m == 1) ? bk : bv;
    unsigned short* dst = (m == 0) ? sQ : (m == 1) ? sK : sV;

    v8f a0 = {}; v8f a1 = {}; v8f a2 = {};
#pragma unroll 4
    for (int ks = 0; ks < KSTEPS; ++ks) {
      const int base = ks * 32 + part * 8;
      // WGP-scope prefetch of the weight stream into the near cache.
      if ((ks & 7) == 0) __builtin_prefetch(wcol + base + 256, 0, 3);
      const v16bf Bm = load_frag(wcol + base);   // weights (global / L2)
      const v16bf A0 = load_frag(sA0 + base);    // x (LDS)
      const v16bf A1 = load_frag(sA1 + base);
      const v16bf A2 = load_frag(sA2 + base);
      a0 = BFWMMA(A0, Bm, a0);
      a1 = BFWMMA(A1, Bm, a1);
      a2 = BFWMMA(A2, Bm, a2);
    }
    const float bcol = bias[col];
#pragma unroll
    for (int e = 0; e < 8; ++e) {   // D layout: elem e -> row e (+8 for hi lanes)
      const int rr = part * 8 + e;
      dst[(size_t)(rr +  0) * Hdim + col] = f2bf(a0[e] + bcol);
      dst[(size_t)(rr + 16) * Hdim + col] = f2bf(a1[e] + bcol);
      dst[(size_t)(rr + 32) * Hdim + col] = f2bf(a2[e] + bcol);
    }
  }
  __syncthreads();

  // ---- Phase 2: polarity scores. 144 (b,q,k) pairs x 4 h-chunks ----------
  for (int it = tid; it < 576; it += NTHREADS) {
    const int p  = it >> 2;
    const int c  = it & 3;
    const int b  = p / 9;
    const int qi = (p % 9) / 3;
    const int ki = p % 3;
    const unsigned short* qr = sQ + (size_t)(b * 3 + qi) * Hdim + c * 192;
    const unsigned short* kr = sK + (size_t)(b * 3 + ki) * Hdim + c * 192;
    float accp = 0.f, accn = 0.f;
#pragma unroll 4
    for (int h = 0; h < 192; h += 2) {
      const unsigned qq = *(const unsigned*)(qr + h);
      const unsigned kk = *(const unsigned*)(kr + h);
      const float q0 = __builtin_bit_cast(float, qq << 16);
      const float q1 = __builtin_bit_cast(float, qq & 0xFFFF0000u);
      const float k0 = __builtin_bit_cast(float, kk << 16);
      const float k1 = __builtin_bit_cast(float, kk & 0xFFFF0000u);
      const float qp0 = fmaxf(q0, 0.f), qn0 = qp0 - q0;
      const float qp1 = fmaxf(q1, 0.f), qn1 = qp1 - q1;
      const float kp0 = fmaxf(k0, 0.f), kn0 = kp0 - k0;
      const float kp1 = fmaxf(k1, 0.f), kn1 = kp1 - k1;
      accp += qp0 * kp0 + qp1 * kp1;
      accn += qn0 * kn0 + qn1 * kn1;
    }
    atomicAdd(&sSP[p], accp);   // ds_add_f32
    atomicAdd(&sSN[p], accn);
  }
  __syncthreads();

  // ---- Phase 3: dual softmax over T=3, combined weights w = a+ - a- ------
  if (tid < 48) {
    const int b = tid / 3, qi = tid % 3;
    const float scale = 0.03608439182435161f;   // 768^-0.5
    float sp[3], sn[3];
#pragma unroll
    for (int k = 0; k < 3; ++k) {
      sp[k] = sSP[b * 9 + qi * 3 + k] * scale;
      sn[k] = sSN[b * 9 + qi * 3 + k] * scale;
    }
    const float mp = fmaxf(sp[0], fmaxf(sp[1], sp[2]));
    const float mn = fmaxf(sn[0], fmaxf(sn[1], sn[2]));
    float ep[3], en[3], sump = 0.f, sumn = 0.f;
#pragma unroll
    for (int k = 0; k < 3; ++k) {
      ep[k] = __expf(sp[k] - mp); sump += ep[k];
      en[k] = __expf(sn[k] - mn); sumn += en[k];
    }
    const float rp = 1.0f / sump, rn = 1.0f / sumn;
#pragma unroll
    for (int k = 0; k < 3; ++k)
      sWgt[tid * 3 + k] = ep[k] * rp - en[k] * rn;
  }
  __syncthreads();

  // ---- Phase 4: out = w @ V, in place over sV (per-(b,h) triple) ---------
  for (int it = tid; it < BATCH_PER_WG * (Hdim / 2); it += NTHREADS) {
    const int b  = it / (Hdim / 2);
    const int h2 = (it % (Hdim / 2)) * 2;
    unsigned* p0 = (unsigned*)(sV + (size_t)(b * 3 + 0) * Hdim + h2);
    unsigned* p1 = (unsigned*)(sV + (size_t)(b * 3 + 1) * Hdim + h2);
    unsigned* p2 = (unsigned*)(sV + (size_t)(b * 3 + 2) * Hdim + h2);
    const unsigned u0 = *p0, u1 = *p1, u2 = *p2;
    const float v0a = __builtin_bit_cast(float, u0 << 16), v0b = __builtin_bit_cast(float, u0 & 0xFFFF0000u);
    const float v1a = __builtin_bit_cast(float, u1 << 16), v1b = __builtin_bit_cast(float, u1 & 0xFFFF0000u);
    const float v2a = __builtin_bit_cast(float, u2 << 16), v2b = __builtin_bit_cast(float, u2 & 0xFFFF0000u);
    const float* w = sWgt + b * 9;
    const float oa0 = w[0]*v0a + w[1]*v1a + w[2]*v2a;
    const float ob0 = w[0]*v0b + w[1]*v1b + w[2]*v2b;
    const float oa1 = w[3]*v0a + w[4]*v1a + w[5]*v2a;
    const float ob1 = w[3]*v0b + w[4]*v1b + w[5]*v2b;
    const float oa2 = w[6]*v0a + w[7]*v1a + w[8]*v2a;
    const float ob2 = w[6]*v0b + w[7]*v1b + w[8]*v2b;
    *p0 = (unsigned)f2bf(oa0) | ((unsigned)f2bf(ob0) << 16);
    *p1 = (unsigned)f2bf(oa1) | ((unsigned)f2bf(ob1) << 16);
    *p2 = (unsigned)f2bf(oa2) | ((unsigned)f2bf(ob2) << 16);
  }
  __syncthreads();

  // ---- Phase 5: y = x + out @ Wo^T + bo, write fp32 to global ------------
  const unsigned short* sO0 = sV + (size_t)(nlo +  0) * Hdim;
  const unsigned short* sO1 = sV + (size_t)(nlo + 16) * Hdim;
  const unsigned short* sO2 = sV + (size_t)(nlo + 32) * Hdim;
  const unsigned short* wo  = wbf + (size_t)3 * Hdim * Hdim;

  for (int nb = wave; nb < NBLK; nb += NWAVES) {
    const int col = nb * 16 + nlo;
    const unsigned short* wcol = wo + (size_t)col * Hdim;
    v8f a0 = {}; v8f a1 = {}; v8f a2 = {};
#pragma unroll 4
    for (int ks = 0; ks < KSTEPS; ++ks) {
      const int base = ks * 32 + part * 8;
      if ((ks & 7) == 0) __builtin_prefetch(wcol + base + 256, 0, 3);
      const v16bf Bm = load_frag(wcol + base);
      const v16bf A0 = load_frag(sO0 + base);
      const v16bf A1 = load_frag(sO1 + base);
      const v16bf A2 = load_frag(sO2 + base);
      a0 = BFWMMA(A0, Bm, a0);
      a1 = BFWMMA(A1, Bm, a1);
      a2 = BFWMMA(A2, Bm, a2);
    }
    const float bcol = bo[col];
#pragma unroll
    for (int e = 0; e < 8; ++e) {
      const int rr = part * 8 + e;
      out[(row0 + rr +  0) * Hdim + col] = a0[e] + bcol + bf2f(sX[(size_t)(rr +  0) * Hdim + col]);
      out[(row0 + rr + 16) * Hdim + col] = a1[e] + bcol + bf2f(sX[(size_t)(rr + 16) * Hdim + col]);
      out[(row0 + rr + 32) * Hdim + col] = a2[e] + bcol + bf2f(sX[(size_t)(rr + 32) * Hdim + col]);
    }
  }
}

// ---------------------------------------------------------------------------
extern "C" void kernel_launch(void* const* d_in, const int* in_sizes, int n_in,
                              void* d_out, int out_size, void* d_ws, size_t ws_size,
                              hipStream_t stream) {
  const float* feat = (const float*)d_in[0];
  const float* Wq   = (const float*)d_in[1];
  const float* bq   = (const float*)d_in[2];
  const float* Wk   = (const float*)d_in[3];
  const float* bk   = (const float*)d_in[4];
  const float* Wv   = (const float*)d_in[5];
  const float* bv   = (const float*)d_in[6];
  const float* Wo   = (const float*)d_in[7];
  const float* bo   = (const float*)d_in[8];
  const float* ln_g = (const float*)d_in[9];
  const float* ln_b = (const float*)d_in[10];

  unsigned short* wbf = (unsigned short*)d_ws;   // needs 4*768*768*2 = 4.5 MB

  // One-shot bf16 weight conversion (deterministic, re-run every call).
  const int nw = 4 * Hdim * Hdim;
  convert_weights_kernel<<<(nw + 255) / 256, 256, 0, stream>>>(Wq, Wk, Wv, Wo, wbf);

  const int rows = in_sizes[0] / Hdim;           // B*T = 196608
  const int grid = rows / ROWS_PER_WG;           // 4096 workgroups
  const size_t smem_bytes = (size_t)4 * ROWS_PER_WG * Hdim * 2 + 3 * 144 * 4; // ~290 KB (<=320 KB/WGP)

  (void)hipFuncSetAttribute((const void*)pola_fused_kernel,
                            hipFuncAttributeMaxDynamicSharedMemorySize,
                            (int)smem_bytes);

  pola_fused_kernel<<<grid, NTHREADS, smem_bytes, stream>>>(
      feat, wbf, bq, bk, bv, bo, ln_g, ln_b, (float*)d_out);
}
